// MultiboxLoss_66090956751428
// MI455X (gfx1250) — compile-verified, matrix operand order checked by reading, not run on hardware
//
#include <hip/hip_runtime.h>
#include <hip/hip_fp16.h>
#include <math.h>

#define B_ 64
#define P_ 24564
#define C_ 81
#define T_ 24
#define TILES ((P_ + 15) / 16)   /* 1536 */
#define STR 82                   /* even LDS row stride: keeps b64-aligned A pairs */

typedef __attribute__((ext_vector_type(2)))  float    v2f;
typedef __attribute__((ext_vector_type(8)))  float    v8f;
typedef __attribute__((ext_vector_type(16))) _Float16 v16h;

// ---------------------------------------------------------------------------
// Kernel 1: per-batch matching. One workgroup per batch image.
// match_idx[b,p] = truth index t if positive, -1 if background.
// overlap>0.5 => best_truth_idx; else forced best-prior override (last t wins).
// ---------------------------------------------------------------------------
__global__ __launch_bounds__(256) void match_kernel(
    const float* __restrict__ targets, const float* __restrict__ anchors,
    int* __restrict__ match) {
  const int b = blockIdx.x, tid = threadIdx.x;
  __shared__ float s_t[T_ * 5];
  __shared__ float s_ov[256];
  __shared__ int   s_pi[256];
  __shared__ int   s_bp[T_];
  if (tid < T_ * 5) s_t[tid] = targets[b * T_ * 5 + tid];
  __syncthreads();

  float bov[T_];
  int   bpi[T_];
#pragma unroll
  for (int t = 0; t < T_; ++t) { bov[t] = -1.0f; bpi[t] = 0; }

  const float4* an4 = (const float4*)anchors;
  for (int p = tid; p < P_; p += 256) {
    const float4 a4 = an4[p];
    const float acx = a4.x, acy = a4.y, aw = a4.z, ah = a4.w;
    const float ax1 = acx - 0.5f * aw, ay1 = acy - 0.5f * ah;
    const float ax2 = acx + 0.5f * aw, ay2 = acy + 0.5f * ah;
    const float area_a = aw * ah;
    float bestov = -1.0f; int bestt = 0;
#pragma unroll
    for (int t = 0; t < T_; ++t) {
      const float tx1 = s_t[t * 5 + 0], ty1 = s_t[t * 5 + 1];
      const float tx2 = s_t[t * 5 + 2], ty2 = s_t[t * 5 + 3];
      const float lx = fmaxf(tx1, ax1), ly = fmaxf(ty1, ay1);
      const float rx = fminf(tx2, ax2), ry = fminf(ty2, ay2);
      const float w = fmaxf(rx - lx, 0.0f), h = fmaxf(ry - ly, 0.0f);
      const float inter = w * h;
      const float area_t = (tx2 - tx1) * (ty2 - ty1);
      const float iou = inter / (area_t + area_a - inter);
      if (iou > bestov) { bestov = iou; bestt = t; }       // first-max tie (t asc)
      if (iou > bov[t]) { bov[t] = iou; bpi[t] = p; }      // lowest p in stride set
    }
    match[b * P_ + p] = (bestov > 0.5f) ? bestt : -1;
  }

  // per-truth argmax over priors (lower p wins ties -> jnp.argmax semantics)
  // fully unrolled so bov/bpi stay in VGPRs (no scratch spill)
#pragma unroll
  for (int t = 0; t < T_; ++t) {
    s_ov[tid] = bov[t]; s_pi[tid] = bpi[t];
    __syncthreads();
    for (int s = 128; s > 0; s >>= 1) {
      if (tid < s) {
        const float o2 = s_ov[tid + s]; const int p2 = s_pi[tid + s];
        if (o2 > s_ov[tid] || (o2 == s_ov[tid] && p2 < s_pi[tid])) {
          s_ov[tid] = o2; s_pi[tid] = p2;
        }
      }
      __syncthreads();
    }
    if (tid == 0) s_bp[t] = s_pi[0];
    __syncthreads();
  }

  if (tid == 0) {
    for (int t = 0; t < T_; ++t) {
      const int ps = s_bp[t];
      bool forcedBefore = false;
      for (int u = 0; u < t; ++u) if (s_bp[u] == ps) forcedBefore = true;
      const int cur = match[b * P_ + ps];
      if (cur < 0 || forcedBefore) match[b * P_ + ps] = t;
    }
  }
}

// ---------------------------------------------------------------------------
// Kernel 2: fused single pass over conf_pred (509 MB dominant stream).
// One wave per 16-prior tile. The 16x81 tile is CONTIGUOUS in memory and its
// base is 16B-aligned, so it is fetched with float4 (global_load_b128) and
// scattered into a stride-82 LDS layout (even stride -> ds_load_2addr_b64
// feeding v_wmma_f32_16x16x4_f32 with a ones B-matrix = exact f32 row sums).
// Emits closs[b,p] (negatives) and posc[b,p] = lse - conf[label] (positives).
// ---------------------------------------------------------------------------
__global__ __launch_bounds__(256) void lse_kernel(
    const float* __restrict__ conf, const float* __restrict__ targets,
    const int* __restrict__ match, float* __restrict__ closs,
    float* __restrict__ posc) {
  __shared__ float sh[8 * (16 * STR + 64)];
  const int tid  = threadIdx.x;
  const int wid  = tid >> 5, lane = tid & 31;
  float* data    = &sh[wid * (16 * STR + 64)];
  float* scratch = data + 16 * STR;      // 32 floats: rowmax partials
  float* rowmax  = scratch + 32;         // 16 floats
  float* rowsum  = rowmax + 16;          // 16 floats

  const int gw   = blockIdx.x * 8 + wid;
  const int b    = gw / TILES;
  const int tile = gw % TILES;
  const int p0   = tile * 16;
  const float* cb = conf + (long)b * P_ * C_;

  // gfx1250 prefetch of next tile (global_prefetch_b8)
  if (p0 + 16 < P_) {
    __builtin_prefetch(cb + (long)(p0 + 16) * C_ + lane * 42, 0, 1);
  }

  // ---- contiguous float4 tile load (tile base always 16B aligned) ----
  const int nrows = (P_ - p0 < 16) ? (P_ - p0) : 16;   // 16 or 4 (P_%16==4)
  const int nf4   = nrows * C_ / 4;                    // 324 or 81 (exact)
  const float4* gb4 = (const float4*)(cb + (long)p0 * C_);
  for (int i = 0; i < 11; ++i) {                       // ceil(324/32)=11
    const int idx4 = i * 32 + lane;
    if (idx4 < 324) {
      float4 v;
      if (idx4 < nf4) v = gb4[idx4];
      else            v = make_float4(0.f, 0.f, 0.f, 0.f);  // zero invalid rows
      const float vv[4] = {v.x, v.y, v.z, v.w};
      const int g = idx4 * 4;
#pragma unroll
      for (int j = 0; j < 4; ++j) {
        const int row = (g + j) / C_, col = (g + j) % C_;
        data[row * STR + col] = vv[j];
      }
    }
  }
  __syncthreads();

  // ---- per-row max, 2 lanes per row ----
  {
    const int row = lane & 15, half = lane >> 4;
    float m = -INFINITY;
    for (int c = half; c < C_; c += 2) m = fmaxf(m, data[row * STR + c]);
    scratch[lane] = m;
  }
  __syncthreads();
  if (lane < 16) rowmax[lane] = fmaxf(scratch[lane], scratch[lane + 16]);
  __syncthreads();

  // ---- capture gathers before exp overwrites the tile ----
  float g0 = 0.0f, gl = 0.0f; int mt = -1;
  const int myp = p0 + lane;
  if (lane < 16 && myp < P_) {
    mt = match[(long)b * P_ + myp];
    const int lbl = (mt >= 0) ? (int)targets[(b * T_ + mt) * 5 + 4] : 0;
    g0 = data[lane * STR + 0];
    gl = data[lane * STR + lbl];
  }
  __syncthreads();

  // ---- exp in place (cols >= 81 untouched/unused) ----
  for (int i = 0; i < (16 * STR) / 32; ++i) {          // 41 iterations
    const int flat = i * 32 + lane;
    const int row = flat / STR, col = flat % STR;
    if (col < C_) data[flat] = expf(data[flat] - rowmax[row]);
  }
  __syncthreads();

  // ---- WMMA row-sum reduction (whole wave, EXEC all ones) ----
#if defined(__has_builtin) && __has_builtin(__builtin_amdgcn_wmma_f32_16x16x4_f32)
  {
    v8f acc = {0.f, 0.f, 0.f, 0.f, 0.f, 0.f, 0.f, 0.f};
    v2f bones; bones.x = 1.0f; bones.y = 1.0f;
    const int m = lane & 15, kb = (lane >> 4) * 2;     // A 16x4 f32 layout
    const float* rowp = &data[m * STR];
#pragma unroll 4
    for (int s = 0; s < 20; ++s) {                     // k = 0..79, all valid
      v2f a; a.x = rowp[4 * s + kb]; a.y = rowp[4 * s + kb + 1];
      acc = __builtin_amdgcn_wmma_f32_16x16x4_f32(false, a, false, bones,
                                                  (short)0, acc, false, false);
    }
    { // tail: k=80 valid, k=81..83 zero
      v2f a; a.x = (kb == 0) ? rowp[80] : 0.0f; a.y = 0.0f;
      acc = __builtin_amdgcn_wmma_f32_16x16x4_f32(false, a, false, bones,
                                                  (short)0, acc, false, false);
    }
    if (lane == 0) {
#pragma unroll
      for (int v = 0; v < 8; ++v) rowsum[v] = acc[v];        // M=0..7,  N=0
    }
    if (lane == 16) {
#pragma unroll
      for (int v = 0; v < 8; ++v) rowsum[8 + v] = acc[v];    // M=8..15, N=0
    }
  }
#else
  {
    v8f acc = {0.f, 0.f, 0.f, 0.f, 0.f, 0.f, 0.f, 0.f};
    v16h bones;
#pragma unroll
    for (int j = 0; j < 16; ++j) bones[j] = (_Float16)1.0f;
    const int m = lane & 15, hi = lane >> 4;           // A 16x32 f16 layout
#pragma unroll
    for (int s = 0; s < 3; ++s) {
      v16h a;
#pragma unroll
      for (int j = 0; j < 16; ++j) {
        const int K = 32 * s + j + 8 * hi + ((j >= 8) ? 8 : 0);
        a[j] = (K < C_) ? (_Float16)data[m * STR + K] : (_Float16)0.0f;
      }
      acc = __builtin_amdgcn_wmma_f32_16x16x32_f16(false, a, false, bones,
                                                   (short)0, acc, false, false);
    }
    if (lane == 0) {
#pragma unroll
      for (int v = 0; v < 8; ++v) rowsum[v] = acc[v];
    }
    if (lane == 16) {
#pragma unroll
      for (int v = 0; v < 8; ++v) rowsum[8 + v] = acc[v];
    }
  }
#endif
  __syncthreads();

  if (lane < 16 && myp < P_) {
    const float lse = rowmax[lane] + logf(rowsum[lane]);
    const bool pos = (mt >= 0);
    closs[(long)b * P_ + myp] = pos ? 0.0f : (lse - g0);  // label 0 = background
    posc [(long)b * P_ + myp] = pos ? (lse - gl) : 0.0f;
  }
}

// ---------------------------------------------------------------------------
// Kernel 3: per-batch hard-negative mining via 4-pass radix select (top-K),
// exact stable ties (index-ascending) via block-wide prefix scan.
// Outputs per batch: [neg_sum, n_neg_excl_pos, n_pos, 0]
// ---------------------------------------------------------------------------
__device__ __forceinline__ unsigned key_of(float f) {
  const unsigned u = __float_as_uint(f);
  return (u & 0x80000000u) ? ~u : (u | 0x80000000u);  // order-preserving map
}

__global__ __launch_bounds__(256) void select_kernel(
    const float* __restrict__ closs, const int* __restrict__ match,
    float* __restrict__ selOut) {
  const int b = blockIdx.x, tid = threadIdx.x;
  const float* cl = closs + (long)b * P_;
  const int*   mt = match + (long)b * P_;
  __shared__ unsigned hist[256];
  __shared__ int   s_red[256];
  __shared__ float s_redf[256];
  __shared__ unsigned s_pref;
  __shared__ int s_rem, s_taken;

  int cnt = 0;
  for (int p = tid; p < P_; p += 256) cnt += (mt[p] >= 0) ? 1 : 0;
  s_red[tid] = cnt; __syncthreads();
  for (int s = 128; s > 0; s >>= 1) {
    if (tid < s) s_red[tid] += s_red[tid + s];
    __syncthreads();
  }
  const int npos = s_red[0];
  __syncthreads();
  const int K = (int)fminf(3.0f * (float)npos, (float)(P_ - 1));

  float negsum = 0.0f; int nneg = 0;
  if (K > 0) {
    unsigned pref = 0; int rem = K;
    for (int pass = 0; pass < 4; ++pass) {
      hist[tid] = 0; __syncthreads();
      const int shift = 24 - 8 * pass;
      for (int p = tid; p < P_; p += 256) {
        const unsigned key = key_of(cl[p]);
        const bool ok = (pass == 0) || ((key >> (shift + 8)) == pref);
        if (ok) atomicAdd(&hist[(key >> shift) & 255u], 1u);
      }
      __syncthreads();
      if (tid == 0) {
        unsigned c = 0; int d = 255;
        while (d > 0 && c + hist[d] < (unsigned)rem) { c += hist[d]; --d; }
        s_pref = (pref << 8) | (unsigned)d;
        s_rem  = rem - (int)c;
      }
      __syncthreads();
      pref = s_pref; rem = s_rem;
      __syncthreads();
    }
    const unsigned kth = pref;
    const int r = rem;  // ties at kth to include, index-ascending (stable sort)

    for (int p = tid; p < P_; p += 256) {
      if (key_of(cl[p]) > kth) { negsum += cl[p]; if (mt[p] < 0) ++nneg; }
    }

    if (tid == 0) s_taken = 0;
    for (int base = 0; base < P_; base += 256) {
      __syncthreads();
      const int taken = s_taken;
      if (taken >= r) break;
      const int p = base + tid;
      const bool eq = (p < P_) && (key_of(cl[p]) == kth);
      s_red[tid] = eq ? 1 : 0;
      __syncthreads();
      for (int off = 1; off < 256; off <<= 1) {
        const int v = (tid >= off) ? s_red[tid - off] : 0;
        __syncthreads();
        s_red[tid] += v;
        __syncthreads();
      }
      const int excl = s_red[tid] - (eq ? 1 : 0);
      const int tot  = s_red[255];
      if (eq && (taken + excl) < r) { negsum += cl[p]; if (mt[p] < 0) ++nneg; }
      __syncthreads();
      if (tid == 0) s_taken = taken + tot;
    }
    __syncthreads();
  }

  s_redf[tid] = negsum; s_red[tid] = nneg; __syncthreads();
  for (int s = 128; s > 0; s >>= 1) {
    if (tid < s) { s_redf[tid] += s_redf[tid + s]; s_red[tid] += s_red[tid + s]; }
    __syncthreads();
  }
  if (tid == 0) {
    selOut[b * 4 + 0] = s_redf[0];
    selOut[b * 4 + 1] = (float)s_red[0];
    selOut[b * 4 + 2] = (float)npos;
    selOut[b * 4 + 3] = 0.0f;
  }
}

// ---------------------------------------------------------------------------
// Kernel 4: smooth-L1 loc loss over positives + positive cls partials,
// deterministic per-block partial sums (no float atomics).
// ---------------------------------------------------------------------------
__global__ __launch_bounds__(256) void posloss_kernel(
    const float* __restrict__ loc, const float* __restrict__ targets,
    const float* __restrict__ anchors, const int* __restrict__ match,
    const float* __restrict__ posc, float* __restrict__ locPart,
    float* __restrict__ poscPart) {
  const int idx = blockIdx.x * 256 + threadIdx.x;  // grid covers B*P exactly
  const int b = idx / P_, p = idx % P_;
  float ll = 0.0f;
  const float pc = posc[idx];
  const int t = match[idx];
  if (t >= 0) {
    const float* tb = &targets[(b * T_ + t) * 5];
    const float cx = 0.5f * (tb[0] + tb[2]), cy = 0.5f * (tb[1] + tb[3]);
    const float w = tb[2] - tb[0], h = tb[3] - tb[1];
    const float4 a4 = ((const float4*)anchors)[p];
    const float e0 = (cx - a4.x) / a4.z, e1 = (cy - a4.y) / a4.w;
    const float e2 = logf(w) - logf(a4.z), e3 = logf(h) - logf(a4.w);
    const float4 lp = ((const float4*)loc)[idx];
    const float d0 = lp.x - e0, d1 = lp.y - e1, d2 = lp.z - e2, d3 = lp.w - e3;
#define SL1_(d) ((fabsf(d) < 1.0f) ? 0.5f * (d) * (d) : fabsf(d) - 0.5f)
    ll = SL1_(d0) + SL1_(d1) + SL1_(d2) + SL1_(d3);
#undef SL1_
  }
  __shared__ float s_a[256], s_b[256];
  s_a[threadIdx.x] = ll; s_b[threadIdx.x] = pc; __syncthreads();
  for (int s = 128; s > 0; s >>= 1) {
    if (threadIdx.x < s) {
      s_a[threadIdx.x] += s_a[threadIdx.x + s];
      s_b[threadIdx.x] += s_b[threadIdx.x + s];
    }
    __syncthreads();
  }
  if (threadIdx.x == 0) { locPart[blockIdx.x] = s_a[0]; poscPart[blockIdx.x] = s_b[0]; }
}

// ---------------------------------------------------------------------------
// Kernel 5: deterministic final reduction -> (loc_loss/N, cls_loss/N)
// ---------------------------------------------------------------------------
__global__ __launch_bounds__(256) void finalize_kernel(
    const float* __restrict__ locPart, const float* __restrict__ poscPart,
    const float* __restrict__ selOut, int nPart, float* __restrict__ out) {
  __shared__ float s_l[256], s_p[256];
  const int tid = threadIdx.x;
  float l = 0.0f, pp = 0.0f;
  for (int i = tid; i < nPart; i += 256) { l += locPart[i]; pp += poscPart[i]; }
  s_l[tid] = l; s_p[tid] = pp; __syncthreads();
  for (int s = 128; s > 0; s >>= 1) {
    if (tid < s) { s_l[tid] += s_l[tid + s]; s_p[tid] += s_p[tid + s]; }
    __syncthreads();
  }
  if (tid == 0) {
    float negsum = 0.0f, nneg = 0.0f, npos = 0.0f;
    for (int bb = 0; bb < B_; ++bb) {
      negsum += selOut[bb * 4 + 0];
      nneg   += selOut[bb * 4 + 1];
      npos   += selOut[bb * 4 + 2];
    }
    const float cls = s_p[0] + negsum +
        ((float)B_ * (float)P_ - npos - nneg) * logf((float)C_);
    out[0] = s_l[0] / npos;
    out[1] = cls / npos;
  }
}

// ---------------------------------------------------------------------------
extern "C" void kernel_launch(void* const* d_in, const int* in_sizes, int n_in,
                              void* d_out, int out_size, void* d_ws, size_t ws_size,
                              hipStream_t stream) {
  (void)in_sizes; (void)n_in; (void)out_size; (void)ws_size;
  const float* loc     = (const float*)d_in[0];
  const float* conf    = (const float*)d_in[1];
  const float* targets = (const float*)d_in[2];
  const float* anchors = (const float*)d_in[3];
  float* out = (float*)d_out;

  const long BP = (long)B_ * P_;              // 1,572,096 (divisible by 256)
  char* w = (char*)d_ws;
  int*   match   = (int*)w;   w += BP * sizeof(int);
  float* closs   = (float*)w; w += BP * sizeof(float);
  float* posc    = (float*)w; w += BP * sizeof(float);
  float* selOut  = (float*)w; w += (size_t)B_ * 4 * sizeof(float);
  const int NPART = (int)(BP / 256);          // 6141
  float* locPart  = (float*)w; w += (size_t)NPART * sizeof(float);
  float* poscPart = (float*)w;

  match_kernel<<<B_, 256, 0, stream>>>(targets, anchors, match);
  const int lseGrid = (B_ * TILES) / 8;       // 8 waves (tiles) per block
  lse_kernel<<<lseGrid, 256, 0, stream>>>(conf, targets, match, closs, posc);
  select_kernel<<<B_, 256, 0, stream>>>(closs, match, selOut);
  posloss_kernel<<<NPART, 256, 0, stream>>>(loc, targets, anchors, match, posc,
                                            locPart, poscPart);
  finalize_kernel<<<1, 256, 0, stream>>>(locPart, poscPart, selOut, NPART, out);
}
